// SelfAttentionBlock_64372969832563
// MI455X (gfx1250) — 16-bit and 8-bit matrix operand order repaired and checked by reading, compiled, not run on hardware
//
#include <hip/hip_runtime.h>

#define DMODEL   1024
#define NHEADS   16
#define DHEAD    64
#define NBATCH   2
#define LTOK     2048
#define NTOK     (NBATCH * LTOK)
#define QKVROWS  (3 * DMODEL)

typedef __attribute__((ext_vector_type(16))) _Float16 v16h;
typedef __attribute__((ext_vector_type(8)))  float    v8f;
typedef __attribute__((ext_vector_type(8)))  int      v8i;

union U16h { v16h v; int4 i[2]; };

__global__ void k_ada(const float* __restrict__ cond, const float* __restrict__ norm_w,
                      float* __restrict__ ada) {
  int idx = blockIdx.x * blockDim.x + threadIdx.x;
  if (idx >= NBATCH * DMODEL) return;
  int nb = idx >> 10, d = idx & 1023;
  const float4* c4 = (const float4*)(cond + (size_t)nb * DMODEL);
  const float4* w4 = (const float4*)(norm_w + (size_t)d * DMODEL);
  float s = 0.f;
  for (int j = 0; j < DMODEL / 4; ++j) {
    float4 a = c4[j], b = w4[j];
    s += a.x * b.x + a.y * b.y + a.z * b.z + a.w * b.w;
  }
  ada[idx] = 1.0f + s;
}

__global__ void k_abs_partial(const float* __restrict__ w, int n, float* __restrict__ part) {
  __shared__ float sm[256];
  float s = 0.f;
  for (int i = blockIdx.x * 256 + threadIdx.x; i < n; i += gridDim.x * 256) s += fabsf(w[i]);
  sm[threadIdx.x] = s; __syncthreads();
  for (int o = 128; o > 0; o >>= 1) {
    if (threadIdx.x < o) sm[threadIdx.x] += sm[threadIdx.x + o];
    __syncthreads();
  }
  if (threadIdx.x == 0) part[blockIdx.x] = sm[0];
}

__global__ void k_abs_final(const float* __restrict__ part, int nblk, int n, float* __restrict__ out2) {
  if (threadIdx.x != 0 || blockIdx.x != 0) return;
  float s = 0.f;
  for (int i = 0; i < nblk; ++i) s += part[i];
  float mean = s / (float)n;
  float deq  = fmaxf(mean, 1e-5f);
  out2[0] = 1.0f / deq;
  out2[1] = deq;
}

__global__ void k_wquant(const float* __restrict__ w, int n4, const float* __restrict__ sc,
                         int* __restrict__ wq) {
  int i = blockIdx.x * blockDim.x + threadIdx.x;
  if (i >= n4) return;
  float s = sc[0];
  float4 v = ((const float4*)w)[i];
  int q0 = (int)rintf(v.x * s); q0 = q0 < -1 ? -1 : (q0 > 1 ? 1 : q0);
  int q1 = (int)rintf(v.y * s); q1 = q1 < -1 ? -1 : (q1 > 1 ? 1 : q1);
  int q2 = (int)rintf(v.z * s); q2 = q2 < -1 ? -1 : (q2 > 1 ? 1 : q2);
  int q3 = (int)rintf(v.w * s); q3 = q3 < -1 ? -1 : (q3 > 1 ? 1 : q3);
  wq[i] = (q0 & 255) | ((q1 & 255) << 8) | ((q2 & 255) << 16) | ((q3 & 255) << 24);
}

__global__ void k_rms_actq(const float* __restrict__ x, const float* __restrict__ ada,
                           int* __restrict__ xq, float* __restrict__ ascale) {
  __shared__ float sm[256];
  int T = blockIdx.x, nb = T >> 11, tid = threadIdx.x;
  float4 xv = ((const float4*)(x + (size_t)T * DMODEL))[tid];
  float ss = xv.x * xv.x + xv.y * xv.y + xv.z * xv.z + xv.w * xv.w;
  sm[tid] = ss; __syncthreads();
  for (int o = 128; o > 0; o >>= 1) { if (tid < o) sm[tid] += sm[tid + o]; __syncthreads(); }
  float r = rsqrtf(sm[0] * (1.0f / (float)DMODEL) + 1e-6f);
  __syncthreads();
  float4 av = ((const float4*)(ada + (size_t)nb * DMODEL))[tid];
  float4 xn = make_float4(xv.x * av.x * r, xv.y * av.y * r, xv.z * av.z * r, xv.w * av.w * r);
  float am = fmaxf(fmaxf(fabsf(xn.x), fabsf(xn.y)), fmaxf(fabsf(xn.z), fabsf(xn.w)));
  sm[tid] = am; __syncthreads();
  for (int o = 128; o > 0; o >>= 1) { if (tid < o) sm[tid] = fmaxf(sm[tid], sm[tid + o]); __syncthreads(); }
  float amax = fmaxf(sm[0], 1e-5f);
  float s = 127.0f / amax;
  int q0 = (int)rintf(xn.x * s); q0 = q0 < -128 ? -128 : (q0 > 127 ? 127 : q0);
  int q1 = (int)rintf(xn.y * s); q1 = q1 < -128 ? -128 : (q1 > 127 ? 127 : q1);
  int q2 = (int)rintf(xn.z * s); q2 = q2 < -128 ? -128 : (q2 > 127 ? 127 : q2);
  int q3 = (int)rintf(xn.w * s); q3 = q3 < -128 ? -128 : (q3 > 127 ? 127 : q3);
  xq[((size_t)T * DMODEL >> 2) + tid] = (q0 & 255) | ((q1 & 255) << 8) | ((q2 & 255) << 16) | ((q3 & 255) << 24);
  if (tid == 0) ascale[T] = amax / 127.0f;
}

__global__ void k_actq(const float* __restrict__ x, int* __restrict__ xq, float* __restrict__ ascale) {
  __shared__ float sm[256];
  int T = blockIdx.x, tid = threadIdx.x;
  float4 xv = ((const float4*)(x + (size_t)T * DMODEL))[tid];
  float am = fmaxf(fmaxf(fabsf(xv.x), fabsf(xv.y)), fmaxf(fabsf(xv.z), fabsf(xv.w)));
  sm[tid] = am; __syncthreads();
  for (int o = 128; o > 0; o >>= 1) { if (tid < o) sm[tid] = fmaxf(sm[tid], sm[tid + o]); __syncthreads(); }
  float amax = fmaxf(sm[0], 1e-5f);
  float s = 127.0f / amax;
  int q0 = (int)rintf(xv.x * s); q0 = q0 < -128 ? -128 : (q0 > 127 ? 127 : q0);
  int q1 = (int)rintf(xv.y * s); q1 = q1 < -128 ? -128 : (q1 > 127 ? 127 : q1);
  int q2 = (int)rintf(xv.z * s); q2 = q2 < -128 ? -128 : (q2 > 127 ? 127 : q2);
  int q3 = (int)rintf(xv.w * s); q3 = q3 < -128 ? -128 : (q3 > 127 ? 127 : q3);
  xq[((size_t)T * DMODEL >> 2) + tid] = (q0 & 255) | ((q1 & 255) << 8) | ((q2 & 255) << 16) | ((q3 & 255) << 24);
  if (tid == 0) ascale[T] = amax / 127.0f;
}

__device__ __forceinline__ v8i load_a8(const char* ap) {
  v8i a;
  int2 a0 = *(const int2*)(ap);
  int2 a1 = *(const int2*)(ap + 16);
  int2 a2 = *(const int2*)(ap + 32);
  int2 a3 = *(const int2*)(ap + 48);
  a[0] = a0.x; a[1] = a0.y; a[2] = a1.x; a[3] = a1.y;
  a[4] = a2.x; a[5] = a2.y; a[6] = a3.x; a[7] = a3.y;
  return a;
}
__device__ __forceinline__ v8i load_b8(const char* bp) {
  v8i b;
  int2 b0 = *(const int2*)(bp);
  int2 b1 = *(const int2*)(bp + 16);
  int2 b2 = *(const int2*)(bp + 32);
  int2 b3 = *(const int2*)(bp + 48);
  b[0] = b0.x; b[1] = b0.y; b[2] = b1.x; b[3] = b1.y;
  b[4] = b2.x; b[5] = b2.y; b[6] = b3.x; b[7] = b3.y;
  return b;
}

__global__ void __launch_bounds__(256) k_qkv_gemm(
    const char* __restrict__ xq, const float* __restrict__ xascale,
    const char* __restrict__ wq, const float* __restrict__ wsc,
    float* __restrict__ qf, float* __restrict__ kf, _Float16* __restrict__ vth) {
  int wid  = (blockIdx.x * blockDim.x + threadIdx.x) >> 5;
  int lane = threadIdx.x & 31;
  int tM  = wid & 255;
  int tNg = wid >> 8;
  int ln = lane & 15;
  bool hiL = lane >= 16;
  const char* ap  = xq + (size_t)(tM * 16 + ln) * DMODEL + (hiL ? 8 : 0);
  const char* bp0 = wq + (size_t)(tNg * 64 + ln) * DMODEL + (hiL ? 8 : 0);
  v8i acc0 = {0, 0, 0, 0, 0, 0, 0, 0};
  v8i acc1 = acc0, acc2 = acc0, acc3 = acc0;
  #pragma unroll 1
  for (int k0 = 0; k0 < DMODEL; k0 += 64) {
    v8i a  = load_a8(ap + k0);
    v8i b0 = load_b8(bp0 + k0);
    v8i b1 = load_b8(bp0 + 16 * DMODEL + k0);
    v8i b2 = load_b8(bp0 + 32 * DMODEL + k0);
    v8i b3 = load_b8(bp0 + 48 * DMODEL + k0);
    acc0 = __builtin_amdgcn_wmma_i32_16x16x64_iu8(true, a, true, b0, acc0, false, false);
    acc1 = __builtin_amdgcn_wmma_i32_16x16x64_iu8(true, a, true, b1, acc1, false, false);
    acc2 = __builtin_amdgcn_wmma_i32_16x16x64_iu8(true, a, true, b2, acc2, false, false);
    acc3 = __builtin_amdgcn_wmma_i32_16x16x64_iu8(true, a, true, b3, acc3, false, false);
  }
  float sw = wsc[1];
  int fbase = tNg * 64;
  int which = fbase >> 10, hf = (fbase >> 6) & 15;
  int nb = (tM * 16) >> 11;
  int rbase = tM * 16 + (hiL ? 8 : 0);
  float asc[8];
  #pragma unroll
  for (int r = 0; r < 8; ++r) asc[r] = xascale[rbase + r] * sw;
  v8i accs[4] = {acc0, acc1, acc2, acc3};
  if (which == 0) {
    float* dst = qf + ((size_t)nb * NHEADS + hf) * LTOK * DHEAD;
    #pragma unroll
    for (int jj = 0; jj < 4; ++jj) {
      int e = jj * 16 + ln;
      #pragma unroll
      for (int r = 0; r < 8; ++r) dst[(size_t)((rbase + r) & 2047) * DHEAD + e] = (float)accs[jj][r] * asc[r];
    }
  } else if (which == 1) {
    float* dst = kf + ((size_t)nb * NHEADS + hf) * LTOK * DHEAD;
    #pragma unroll
    for (int jj = 0; jj < 4; ++jj) {
      int e = jj * 16 + ln;
      #pragma unroll
      for (int r = 0; r < 8; ++r) dst[(size_t)((rbase + r) & 2047) * DHEAD + e] = (float)accs[jj][r] * asc[r];
    }
  } else {
    _Float16* dst = vth + ((size_t)nb * NHEADS + hf) * DHEAD * LTOK;
    #pragma unroll
    for (int jj = 0; jj < 4; ++jj) {
      int e = jj * 16 + ln;
      #pragma unroll
      for (int r = 0; r < 8; ++r)
        dst[(size_t)e * LTOK + ((rbase + r) & 2047)] = (_Float16)((float)accs[jj][r] * asc[r]);
    }
  }
}

__global__ void k_rope(const float* __restrict__ qf, const float* __restrict__ kf,
                       const float* __restrict__ pos, const float* __restrict__ scale,
                       const float* __restrict__ freqs,
                       _Float16* __restrict__ qh, _Float16* __restrict__ kh) {
  int flat = blockIdx.x;
  int t  = flat & 2047;
  int h  = (flat >> 11) & 15;
  int nb = flat >> 15;
  int half = threadIdx.x >> 6;
  int e    = threadIdx.x & 63;
  const float* src = (half ? kf : qf) + (((size_t)nb * NHEADS + h) * LTOK + t) * DHEAD;
  float v = src[e];
  __shared__ float red[128];
  __shared__ float buf[2][64];
  red[threadIdx.x] = v * v;
  __syncthreads();
  for (int o = 32; o > 0; o >>= 1) {
    if (e < o) red[threadIdx.x] += red[threadIdx.x + o];
    __syncthreads();
  }
  float ssum = red[half * 64];
  float fac = sqrtf(scale[h]) * rsqrtf(ssum + 1e-6f);
  float vn = v * fac;
  buf[half][e] = vn;
  __syncthreads();
  float outv;
  if (e < 48) {
    int ee = (e < 24) ? e : (e - 24);
    int a = ee >> 3, fi = ee & 7;
    float th = pos[((size_t)nb * LTOK + t) * 3 + a] * freqs[h * 8 + fi];
    float c = __cosf(th), s = __sinf(th);
    outv = (e < 24) ? (vn * c - buf[half][e + 24] * s)
                    : (vn * c + buf[half][e - 24] * s);
  } else {
    outv = vn;
  }
  _Float16* dst = (half ? kh : qh) + (((size_t)nb * NHEADS + h) * LTOK + t) * DHEAD;
  dst[e] = (_Float16)outv;
}

#define KT_PITCH 72
#define VT_PITCH 40
__global__ void __launch_bounds__(256) k_attn(const _Float16* __restrict__ qh,
                                              const _Float16* __restrict__ kh,
                                              const _Float16* __restrict__ vth,
                                              float* __restrict__ o) {
  __shared__ _Float16 kT[32 * KT_PITCH];
  __shared__ _Float16 vT[64 * VT_PITCH];
  int tid  = threadIdx.x;
  int lane = tid & 31;
  int wv   = tid >> 5;
  int qg = blockIdx.x & 15;
  int h  = (blockIdx.x >> 4) & 15;
  int nb = blockIdx.x >> 8;
  int qt = qg * 8 + wv;
  int ln = lane & 15;
  bool hiL = lane >= 16;
  int db   = hiL ? 8 : 0;
  int voff = hiL ? 8 : 0;

  const _Float16* qbase = qh + (((size_t)nb * NHEADS + h) * LTOK) * DHEAD;
  const _Float16* kbase = kh + (((size_t)nb * NHEADS + h) * LTOK) * DHEAD;
  const _Float16* vbase = vth + (((size_t)nb * NHEADS + h) * DHEAD) * LTOK;

  const _Float16* qp = qbase + (size_t)(qt * 16 + ln) * DHEAD + (hiL ? 8 : 0);
  U16h bqlo, bqhi;
  bqlo.i[0] = *(const int4*)(qp);
  bqlo.i[1] = *(const int4*)(qp + 16);
  bqhi.i[0] = *(const int4*)(qp + 32);
  bqhi.i[1] = *(const int4*)(qp + 48);

  int krow = tid >> 3, kcol = (tid & 7) * 8;
  int vrow = tid >> 2, vcol = (tid & 3) * 8;

  float mrun = -3.0e30f;
  float lrun = 0.0f;
  v8f zero = {0, 0, 0, 0, 0, 0, 0, 0};
  v8f acc0 = zero, acc1 = zero, acc2 = zero, acc3 = zero;

  #pragma unroll 1
  for (int j = 0; j < LTOK; j += 32) {
    __syncthreads();
    *(int4*)&kT[krow * KT_PITCH + kcol] =
        *(const int4*)(kbase + (size_t)(j + krow) * DHEAD + kcol);
    *(int4*)&vT[vrow * VT_PITCH + vcol] =
        *(const int4*)(vbase + (size_t)vrow * LTOK + j + vcol);
    __syncthreads();

    const _Float16* kp0 = &kT[ln * KT_PITCH + db];
    const _Float16* kp1 = &kT[(16 + ln) * KT_PITCH + db];
    U16h a0lo, a0hi, a1lo, a1hi;
    a0lo.i[0] = *(const int4*)(kp0);      a0lo.i[1] = *(const int4*)(kp0 + 16);
    a0hi.i[0] = *(const int4*)(kp0 + 32); a0hi.i[1] = *(const int4*)(kp0 + 48);
    a1lo.i[0] = *(const int4*)(kp1);      a1lo.i[1] = *(const int4*)(kp1 + 16);
    a1hi.i[0] = *(const int4*)(kp1 + 32); a1hi.i[1] = *(const int4*)(kp1 + 48);

    v8f s0 = zero, s1 = zero;
    s0 = __builtin_amdgcn_wmma_f32_16x16x32_f16(false, a0lo.v, false, bqlo.v, (short)0, s0, false, false);
    s0 = __builtin_amdgcn_wmma_f32_16x16x32_f16(false, a0hi.v, false, bqhi.v, (short)0, s0, false, false);
    s1 = __builtin_amdgcn_wmma_f32_16x16x32_f16(false, a1lo.v, false, bqlo.v, (short)0, s1, false, false);
    s1 = __builtin_amdgcn_wmma_f32_16x16x32_f16(false, a1hi.v, false, bqhi.v, (short)0, s1, false, false);

    float smax = s0[0];
    #pragma unroll
    for (int r = 1; r < 8; ++r) smax = fmaxf(smax, s0[r]);
    #pragma unroll
    for (int r = 0; r < 8; ++r) smax = fmaxf(smax, s1[r]);
    smax = fmaxf(smax, __shfl_xor(smax, 16, 32));
    float mnew  = fmaxf(mrun, smax);
    float alpha = __expf(mrun - mnew);
    v16h bp;
    float ps = 0.f;
    #pragma unroll
    for (int r = 0; r < 8; ++r) { float p = __expf(s0[r] - mnew); ps += p; bp[r] = (_Float16)p; }
    #pragma unroll
    for (int r = 0; r < 8; ++r) { float p = __expf(s1[r] - mnew); ps += p; bp[8 + r] = (_Float16)p; }
    ps += __shfl_xor(ps, 16, 32);
    lrun = lrun * alpha + ps;
    mrun = mnew;
    #pragma unroll
    for (int r = 0; r < 8; ++r) {
      acc0[r] *= alpha; acc1[r] *= alpha; acc2[r] *= alpha; acc3[r] *= alpha;
    }

    U16h av0, av1, av2, av3;
    {
      const _Float16* v0 = &vT[(0 * 16 + ln) * VT_PITCH + voff];
      const _Float16* v1 = &vT[(1 * 16 + ln) * VT_PITCH + voff];
      const _Float16* v2 = &vT[(2 * 16 + ln) * VT_PITCH + voff];
      const _Float16* v3 = &vT[(3 * 16 + ln) * VT_PITCH + voff];
      av0.i[0] = *(const int4*)(v0); av0.i[1] = *(const int4*)(v0 + 16);
      av1.i[0] = *(const int4*)(v1); av1.i[1] = *(const int4*)(v1 + 16);
      av2.i[0] = *(const int4*)(v2); av2.i[1] = *(const int4*)(v2 + 16);
      av3.i[0] = *(const int4*)(v3); av3.i[1] = *(const int4*)(v3 + 16);
    }
    acc0 = __builtin_amdgcn_wmma_f32_16x16x32_f16(false, av0.v, false, bp, (short)0, acc0, false, false);
    acc1 = __builtin_amdgcn_wmma_f32_16x16x32_f16(false, av1.v, false, bp, (short)0, acc1, false, false);
    acc2 = __builtin_amdgcn_wmma_f32_16x16x32_f16(false, av2.v, false, bp, (short)0, acc2, false, false);
    acc3 = __builtin_amdgcn_wmma_f32_16x16x32_f16(false, av3.v, false, bp, (short)0, acc3, false, false);
  }

  float inv = 1.0f / lrun;
  float* op = o + (size_t)(nb * LTOK + qt * 16 + ln) * DMODEL + h * DHEAD + (hiL ? 8 : 0);
  v8f accs[4] = {acc0, acc1, acc2, acc3};
  #pragma unroll
  for (int c = 0; c < 4; ++c) {
    float4 u0 = make_float4(accs[c][0] * inv, accs[c][1] * inv, accs[c][2] * inv, accs[c][3] * inv);
    float4 u1 = make_float4(accs[c][4] * inv, accs[c][5] * inv, accs[c][6] * inv, accs[c][7] * inv);
    *(float4*)(op + c * 16)     = u0;
    *(float4*)(op + c * 16 + 4) = u1;
  }
}

__global__ void __launch_bounds__(256) k_out_gemm(
    const char* __restrict__ oq, const float* __restrict__ oascale,
    const char* __restrict__ wq, const float* __restrict__ wsc,
    const float* __restrict__ skip, float* __restrict__ out) {
  int wid  = (blockIdx.x * blockDim.x + threadIdx.x) >> 5;
  int lane = threadIdx.x & 31;
  int tM  = wid & 255;
  int tNg = wid >> 8;
  int ln = lane & 15;
  bool hiL = lane >= 16;
  const char* ap  = oq + (size_t)(tM * 16 + ln) * DMODEL + (hiL ? 8 : 0);
  const char* bp0 = wq + (size_t)(tNg * 64 + ln) * DMODEL + (hiL ? 8 : 0);
  v8i acc0 = {0, 0, 0, 0, 0, 0, 0, 0};
  v8i acc1 = acc0, acc2 = acc0, acc3 = acc0;
  #pragma unroll 1
  for (int k0 = 0; k0 < DMODEL; k0 += 64) {
    v8i a  = load_a8(ap + k0);
    v8i b0 = load_b8(bp0 + k0);
    v8i b1 = load_b8(bp0 + 16 * DMODEL + k0);
    v8i b2 = load_b8(bp0 + 32 * DMODEL + k0);
    v8i b3 = load_b8(bp0 + 48 * DMODEL + k0);
    acc0 = __builtin_amdgcn_wmma_i32_16x16x64_iu8(true, a, true, b0, acc0, false, false);
    acc1 = __builtin_amdgcn_wmma_i32_16x16x64_iu8(true, a, true, b1, acc1, false, false);
    acc2 = __builtin_amdgcn_wmma_i32_16x16x64_iu8(true, a, true, b2, acc2, false, false);
    acc3 = __builtin_amdgcn_wmma_i32_16x16x64_iu8(true, a, true, b3, acc3, false, false);
  }
  float sw = wsc[1];
  int rbase = tM * 16 + (hiL ? 8 : 0);
  float asc[8];
  #pragma unroll
  for (int r = 0; r < 8; ++r) asc[r] = oascale[rbase + r] * sw;
  v8i accs[4] = {acc0, acc1, acc2, acc3};
  #pragma unroll
  for (int jj = 0; jj < 4; ++jj) {
    int f = tNg * 64 + jj * 16 + ln;
    #pragma unroll
    for (int r = 0; r < 8; ++r) {
      int t = rbase + r;
      out[(size_t)t * DMODEL + f] = (float)accs[jj][r] * asc[r] + skip[(size_t)t * DMODEL + f];
    }
  }
}

extern "C" void kernel_launch(void* const* d_in, const int* in_sizes, int n_in,
                              void* d_out, int out_size, void* d_ws, size_t ws_size,
                              hipStream_t stream) {
  const float* x      = (const float*)d_in[0];
  const float* pos    = (const float*)d_in[1];
  const float* cond   = (const float*)d_in[2];
  const float* norm_w = (const float*)d_in[3];
  const float* qkv_w  = (const float*)d_in[4];
  const float* out_w  = (const float*)d_in[5];
  const float* scale  = (const float*)d_in[6];
  const float* freqs  = (const float*)d_in[7];
  float* out = (float*)d_out;

  char* base = (char*)d_ws;
  size_t off = 0;
  auto take = [&](size_t bytes) -> char* {
    char* p = base + off;
    off += (bytes + 255) & ~(size_t)255;
    return p;
  };
  float*    ada    = (float*)take((size_t)NBATCH * DMODEL * 4);
  float*    wpartQ = (float*)take(256 * 4);
  float*    wpartO = (float*)take(256 * 4);
  float*    wscQ   = (float*)take(256);
  float*    wscO   = (float*)take(256);
  float*    xasc   = (float*)take((size_t)NTOK * 4);
  float*    oasc   = (float*)take((size_t)NTOK * 4);
  char*     wqQ    = (char*)take((size_t)QKVROWS * DMODEL);
  char*     wqO    = (char*)take((size_t)DMODEL * DMODEL);
  char*     xq     = (char*)take((size_t)NTOK * DMODEL);
  _Float16* qhB    = (_Float16*)take((size_t)NTOK * DMODEL * 2);
  _Float16* khB    = (_Float16*)take((size_t)NTOK * DMODEL * 2);
  _Float16* vthB   = (_Float16*)take((size_t)NTOK * DMODEL * 2);
  float*    qf     = (float*)take((size_t)NTOK * DMODEL * 4);
  float*    kf     = (float*)take((size_t)NTOK * DMODEL * 4);
  float*    of     = qf;
  char*     oq     = (char*)kf;

  k_ada<<<8, 256, 0, stream>>>(cond, norm_w, ada);
  k_abs_partial<<<256, 256, 0, stream>>>(qkv_w, QKVROWS * DMODEL, wpartQ);
  k_abs_partial<<<256, 256, 0, stream>>>(out_w, DMODEL * DMODEL, wpartO);
  k_abs_final<<<1, 1, 0, stream>>>(wpartQ, 256, QKVROWS * DMODEL, wscQ);
  k_abs_final<<<1, 1, 0, stream>>>(wpartO, 256, DMODEL * DMODEL, wscO);
  k_wquant<<<(QKVROWS * DMODEL / 4) / 256, 256, 0, stream>>>(qkv_w, QKVROWS * DMODEL / 4, wscQ, (int*)wqQ);
  k_wquant<<<(DMODEL * DMODEL / 4) / 256, 256, 0, stream>>>(out_w, DMODEL * DMODEL / 4, wscO, (int*)wqO);
  k_rms_actq<<<NTOK, 256, 0, stream>>>(x, ada, (int*)xq, xasc);
  k_qkv_gemm<<<(256 * 48) / 8, 256, 0, stream>>>(xq, xasc, wqQ, wscQ, qf, kf, vthB);
  k_rope<<<NBATCH * NHEADS * LTOK, 128, 0, stream>>>(qf, kf, pos, scale, freqs, qhB, khB);
  k_attn<<<NBATCH * NHEADS * 16, 256, 0, stream>>>(qhB, khB, vthB, of);
  k_actq<<<NTOK, 256, 0, stream>>>(of, (int*)oq, oasc);
  k_out_gemm<<<(256 * 16) / 8, 256, 0, stream>>>(oq, oasc, wqO, wscO, x, out);
  (void)in_sizes; (void)n_in; (void)out_size; (void)ws_size;
}
